// GATConvLayer_24163486007666
// MI455X (gfx1250) — matrix operand order repaired and checked by reading, not compiled, not run on hardware
//
#include <hip/hip_runtime.h>
#include <hip/hip_bf16.h>
#include <math.h>

// ---------------- problem constants (match reference) ----------------
#define N_NODES 50000
#define N_EDGES 800000
#define D_IN    128
#define HD      128     // N_HEADS * D_OUT
#define N_HEADS 8
#define D_OUT   16

typedef __bf16 bf16_t;
typedef __attribute__((ext_vector_type(16))) __bf16 v16bf;
typedef __attribute__((ext_vector_type(8)))  float  v8f;
typedef __attribute__((ext_vector_type(4)))  unsigned int u32x4;
typedef __attribute__((ext_vector_type(8)))  int i32x8;
typedef __attribute__((ext_vector_type(4)))  int i32x4;

#define W_ELEMS   16384            // 128*128 bf16 per plane
#define W_PACKED  (2 * W_ELEMS)    // hi plane then lo plane (ushort each)

// -------- monotone int encoding of float for atomicMax over signed ints -----
__device__ __forceinline__ int enc_f32(float f) {
    int i = __float_as_int(f);
    return (i >= 0) ? i : (i ^ 0x7fffffff);
}
__device__ __forceinline__ float dec_f32(int c) {
    int b = (c >= 0) ? c : (c ^ 0x7fffffff);
    return __int_as_float(b);
}
#define ENC_NEG_INF ((int)0x807FFFFF)   // enc(-inf)

__device__ __forceinline__ void atomic_fadd(float* p, float v) {
    __hip_atomic_fetch_add(p, v, __ATOMIC_RELAXED, __HIP_MEMORY_SCOPE_AGENT);
}

// load 16 contiguous bf16 (32B) -> v16bf (two b128 loads; ds_load when in LDS)
__device__ __forceinline__ v16bf ld_bf16x16(const unsigned short* p) {
    struct U { uint4 a, b; };
    U u = *(const U*)p;
    return __builtin_bit_cast(v16bf, u);
}

// ---------------------------------------------------------------------------
// Tensor Data Mover: DMA the 64KB packed-weight buffer (hi+lo planes,
// 16384 dwords contiguous) from global into LDS.  D# per ISA ch.8:
//   group0 = {count=1, lds_addr, global_addr_lo, global_addr_hi | type=2}
//   group1 = 1-D tile: data_size=4B, tensor_dim0=tile_dim0=16384,
//            tensor_dim1=tile_dim1=1, tensor_dim0_stride=16384
// This toolchain exposes the 6-arg builtin form.
// ---------------------------------------------------------------------------
#if defined(__has_builtin)
#if __has_builtin(__builtin_amdgcn_tensor_load_to_lds)
#define HAVE_TDM 1
#endif
#endif

__device__ __forceinline__ void tdm_stage_weights(const unsigned short* g,
                                                  unsigned lds_addr) {
#ifdef HAVE_TDM
    unsigned long long ga = (unsigned long long)(size_t)g;
    u32x4 g0;
    g0[0] = 1u;                                            // count=1, user D#
    g0[1] = lds_addr;                                      // lds_addr (bytes)
    g0[2] = (unsigned)(ga & 0xffffffffu);                  // global_addr[31:0]
    g0[3] = (unsigned)((ga >> 32) & 0x01ffffffu) | (2u << 30); // [56:32] | type=2
    i32x8 g1;
    g1[0] = (int)(2u << 16);        // workgroup_mask=0, data_size=2 (4B)
    g1[1] = (int)(16384u << 16);    // tensor_dim0[15:0] in bits[31:16]
    g1[2] = (int)(1u << 16);        // tensor_dim0[31:16]=0 | tensor_dim1[15:0]=1
    g1[3] = (int)(16384u << 16);    // tensor_dim1[31:16]=0 | tile_dim0=16384
    g1[4] = 1;                      // tile_dim1=1, tile_dim2=0
    g1[5] = 16384;                  // tensor_dim0_stride[31:0]
    g1[6] = 0;                      // stride hi / dim1_stride lo
    g1[7] = 0;
    i32x4 z4 = {0, 0, 0, 0};
    i32x8 z8 = {0, 0, 0, 0, 0, 0, 0, 0};
    __builtin_amdgcn_tensor_load_to_lds(g0, g1, z4, z4, z8, 0);
    __builtin_amdgcn_s_wait_tensorcnt(0);
#else
    (void)g; (void)lds_addr;
#endif
}

// ---------------------------------------------------------------------------
// Pre-split weights:  WT[col*128 + k] = bf16_hi(W[k][col]); plane 1 = bf16_lo.
// Transposed so a B fragment (one output col, 16 consecutive k) is contiguous.
// ---------------------------------------------------------------------------
__global__ void pack_weights_kernel(const float* __restrict__ W,
                                    unsigned short* __restrict__ WT)
{
    const int idx = blockIdx.x * blockDim.x + threadIdx.x;
    if (idx >= W_ELEMS) return;
    const int col = idx >> 7, k = idx & 127;
    const float w = W[(size_t)k * HD + col];
    const bf16_t h = (bf16_t)w;
    const bf16_t l = (bf16_t)(w - (float)h);
    WT[idx]           = __builtin_bit_cast(unsigned short, h);
    WT[W_ELEMS + idx] = __builtin_bit_cast(unsigned short, l);
}

// ---------------------------------------------------------------------------
// Fused [M,128] x [128,128] GEMM, fp32 via bf16x3 on V_WMMA_F32_16X16X32_BF16.
// One wave per 16x16 output tile; 4 waves per block; weights staged in LDS by
// the TDM.  EDGE_MODE epilogue fuses econv's segment_sum with atomic adds.
// ---------------------------------------------------------------------------
template <bool EDGE_MODE>
__global__ __launch_bounds__(128)
void gemm128_bf16x3_kernel(const float* __restrict__ A,          // [M,128]
                           const unsigned short* __restrict__ WT, // packed hi/lo
                           float* __restrict__ C,
                           const int* __restrict__ dstIdx,
                           const float* __restrict__ bias,
                           int Mtiles)
{
    __shared__ unsigned short sW[W_PACKED];   // 64KB: hi plane @0, lo @W_ELEMS

    if (threadIdx.x < 32) {                   // wave 0 drives the TDM
        tdm_stage_weights(WT, (unsigned)(size_t)&sW[0]);
    }
    __syncthreads();

    const int wave     = blockIdx.x * (blockDim.x >> 5) + (threadIdx.x >> 5);
    const int row_tile = wave >> 3;
    const int col_tile = wave & 7;
    if (row_tile >= Mtiles) return;           // grids sized exactly; never taken

    const int lane = threadIdx.x & 31;
    const int half = lane >> 4;
    const int r16  = lane & 15;
    const int row0 = row_tile << 4;
    const int colg = (col_tile << 4) + r16;

    const float* __restrict__ Arow = A + (size_t)(row0 + r16) * D_IN;
    const unsigned short* sBhi = sW + colg * 128;            // [col][k] bf16
    const unsigned short* sBlo = sBhi + W_ELEMS;

    v8f acc = {};

#pragma unroll
    for (int k0 = 0; k0 < D_IN; k0 += 32) {
        // ---- A fragment 16x32 bf16: lane<16 -> K=k0+[0..15]
        //                             lane>=16 -> K=k0+16+[0..15]
        const int kA = k0 + half * 16;
        float av[16];
        ((float4*)av)[0] = *(const float4*)(Arow + kA);
        ((float4*)av)[1] = *(const float4*)(Arow + kA + 4);
        ((float4*)av)[2] = *(const float4*)(Arow + kA + 8);
        ((float4*)av)[3] = *(const float4*)(Arow + kA + 12);
        if (k0 + 32 < D_IN) __builtin_prefetch(Arow + kA + 32, 0, 0);

        v16bf a_hi, a_lo;
#pragma unroll
        for (int i = 0; i < 16; ++i) {
            bf16_t h = (bf16_t)av[i];
            a_hi[i]  = h;
            a_lo[i]  = (bf16_t)(av[i] - (float)h);
        }

        // ---- B fragment 32x16 bf16 from LDS: lane<16 -> K=k0..k0+15,
        //      lane>=16 -> K=k0+16..k0+31 ; contiguous per column
        const int kB = k0 + half * 16;
        v16bf b_hi = ld_bf16x16(sBhi + kB);
        v16bf b_lo = ld_bf16x16(sBlo + kB);

        // ---- bf16x3 fp32 emulation: Ahi*Bhi + Ahi*Blo + Alo*Bhi
        acc = __builtin_amdgcn_wmma_f32_16x16x32_bf16(false, a_hi, false, b_hi,
                                                      (short)0, acc, false, false);
        acc = __builtin_amdgcn_wmma_f32_16x16x32_bf16(false, a_hi, false, b_lo,
                                                      (short)0, acc, false, false);
        acc = __builtin_amdgcn_wmma_f32_16x16x32_bf16(false, a_lo, false, b_hi,
                                                      (short)0, acc, false, false);
    }

    // ---- epilogue.  C layout: VGPR i -> tile row (i + 8*half), col = lane&15
    if (!EDGE_MODE) {
#pragma unroll
        for (int i = 0; i < 8; ++i) {
            const int row = row0 + i + 8 * half;
            C[(size_t)row * HD + colg] = acc[i];
        }
    } else {
        const float be = bias[colg];
#pragma unroll
        for (int i = 0; i < 8; ++i) {
            const int erow = row0 + i + 8 * half;   // edge index
            const int d    = dstIdx[erow];
            atomic_fadd(&C[(size_t)d * HD + colg], acc[i] + be);
        }
    }
}

// ---------------------------------------------------------------------------
__global__ void init_kernel(float* __restrict__ out, float* __restrict__ esum,
                            int* __restrict__ menc, float* __restrict__ denom,
                            float* __restrict__ indeg)
{
    const int idx = blockIdx.x * blockDim.x + threadIdx.x;
    if (idx < N_NODES * HD) { out[idx] = 0.0f; esum[idx] = 0.0f; }
    if (idx < N_NODES * N_HEADS) { menc[idx] = ENC_NEG_INF; denom[idx] = 0.0f; }
    if (idx < N_NODES) indeg[idx] = 0.0f;
}

__global__ void attn_score_kernel(const float* __restrict__ hf,
                                  const float* __restrict__ attn_l,
                                  const float* __restrict__ attn_r,
                                  float* __restrict__ el, float* __restrict__ er)
{
    const int idx = blockIdx.x * blockDim.x + threadIdx.x;
    if (idx >= N_NODES * N_HEADS) return;
    const int n = idx >> 3, h = idx & 7;
    const float* hv = hf + (size_t)n * HD + h * D_OUT;
    const float* al = attn_l + h * D_OUT;
    const float* ar = attn_r + h * D_OUT;
    float sl = 0.0f, sr = 0.0f;
#pragma unroll
    for (int d = 0; d < D_OUT; ++d) {
        const float v = hv[d];
        sl += v * al[d];
        sr += v * ar[d];
    }
    el[idx] = sl;
    er[idx] = sr;
}

__device__ __forceinline__ float edge_score(const float* el, const float* er,
                                            int s, int d, int h)
{
    float x = el[s * N_HEADS + h] + er[d * N_HEADS + h];
    return (x > 0.0f) ? x : 0.2f * x;   // leaky_relu, slope 0.2
}

__global__ void edge_max_kernel(const int* __restrict__ src, const int* __restrict__ dst,
                                const float* __restrict__ el, const float* __restrict__ er,
                                int* __restrict__ menc, float* __restrict__ indeg)
{
    const int idx = blockIdx.x * blockDim.x + threadIdx.x;
    if (idx >= N_EDGES * N_HEADS) return;
    const int e = idx >> 3, h = idx & 7;
    const int s = src[e], d = dst[e];
    const float x = edge_score(el, er, s, d, h);
    atomicMax(&menc[d * N_HEADS + h], enc_f32(x));
    if (h == 0) atomic_fadd(&indeg[d], 1.0f);
}

__global__ void edge_den_kernel(const int* __restrict__ src, const int* __restrict__ dst,
                                const float* __restrict__ el, const float* __restrict__ er,
                                const int* __restrict__ menc, float* __restrict__ denom)
{
    const int idx = blockIdx.x * blockDim.x + threadIdx.x;
    if (idx >= N_EDGES * N_HEADS) return;
    const int e = idx >> 3, h = idx & 7;
    const int s = src[e], d = dst[e];
    const float x  = edge_score(el, er, s, d, h);
    const float mm = dec_f32(menc[d * N_HEADS + h]);
    atomic_fadd(&denom[d * N_HEADS + h], __expf(x - mm));
}

__global__ void aggregate_kernel(const int* __restrict__ src, const int* __restrict__ dst,
                                 const float* __restrict__ el, const float* __restrict__ er,
                                 const int* __restrict__ menc, const float* __restrict__ denom,
                                 const float* __restrict__ hf, float* __restrict__ out)
{
    const long long idx = (long long)blockIdx.x * blockDim.x + threadIdx.x;
    if (idx >= (long long)N_EDGES * HD) return;
    const int e = (int)(idx >> 7);
    const int c = (int)(idx & 127);
    const int h = c >> 4;
    const int s = src[e], d = dst[e];
    const float x  = edge_score(el, er, s, d, h);
    const float mm = dec_f32(menc[d * N_HEADS + h]);
    const float a  = __expf(x - mm) / denom[d * N_HEADS + h];
    atomic_fadd(&out[(size_t)d * HD + c], a * hf[(size_t)s * HD + c]);
}

__global__ void finalize_kernel(float* __restrict__ out, const float* __restrict__ bias,
                                const float* __restrict__ hf, const float* __restrict__ esum,
                                const float* __restrict__ indeg)
{
    const int idx = blockIdx.x * blockDim.x + threadIdx.x;
    if (idx >= N_NODES * HD) return;
    const int n = idx >> 7, c = idx & 127;
    const float deg = indeg[n];
    out[idx] += bias[c] + hf[idx] / (deg + 1.0f) + esum[idx] / fmaxf(deg, 1.0f);
}

// ---------------------------------------------------------------------------
extern "C" void kernel_launch(void* const* d_in, const int* in_sizes, int n_in,
                              void* d_out, int out_size, void* d_ws, size_t ws_size,
                              hipStream_t stream)
{
    const float* nfeat  = (const float*)d_in[0];
    const float* efeat  = (const float*)d_in[1];
    const int*   src    = (const int*)  d_in[2];
    const int*   dst    = (const int*)  d_in[3];
    const float* W_fc   = (const float*)d_in[4];
    const float* attn_l = (const float*)d_in[5];
    const float* attn_r = (const float*)d_in[6];
    const float* bias   = (const float*)d_in[7];
    const float* W_e    = (const float*)d_in[8];
    const float* b_e    = (const float*)d_in[9];
    float* out = (float*)d_out;

    // workspace layout (~55 MB + 128KB packed weights)
    float* ws    = (float*)d_ws;
    float* hf    = ws;                                         // N*128
    float* el    = hf + (size_t)N_NODES * HD;                  // N*8
    float* er    = el + (size_t)N_NODES * N_HEADS;             // N*8
    int*   menc  = (int*)(er + (size_t)N_NODES * N_HEADS);     // N*8
    float* denom = (float*)(menc + (size_t)N_NODES * N_HEADS); // N*8
    float* indeg = denom + (size_t)N_NODES * N_HEADS;          // N
    float* esum  = indeg + N_NODES;                            // N*128
    unsigned short* wfcT = (unsigned short*)(esum + (size_t)N_NODES * HD);
    unsigned short* weT  = wfcT + W_PACKED;

    const int THREADS = 256;
    const int gNC = (N_NODES * HD + THREADS - 1) / THREADS;
    const int gNH = (N_NODES * N_HEADS + THREADS - 1) / THREADS;
    const int gEH = (N_EDGES * N_HEADS + THREADS - 1) / THREADS;
    const long long ec = (long long)N_EDGES * HD;
    const int gEC = (int)((ec + THREADS - 1) / THREADS);
    const int gW  = (W_ELEMS + THREADS - 1) / THREADS;

    const int nodeTiles = (N_NODES / 16) * 8;   // 25000 (divisible by 4)
    const int edgeTiles = (N_EDGES / 16) * 8;   // 400000

    init_kernel<<<gNC, THREADS, 0, stream>>>(out, esum, menc, denom, indeg);
    pack_weights_kernel<<<gW, THREADS, 0, stream>>>(W_fc, wfcT);
    pack_weights_kernel<<<gW, THREADS, 0, stream>>>(W_e, weT);

    gemm128_bf16x3_kernel<false><<<nodeTiles / 4, 128, 0, stream>>>(
        nfeat, wfcT, hf, nullptr, nullptr, N_NODES / 16);

    attn_score_kernel<<<gNH, THREADS, 0, stream>>>(hf, attn_l, attn_r, el, er);

    gemm128_bf16x3_kernel<true><<<edgeTiles / 4, 128, 0, stream>>>(
        efeat, weT, esum, dst, b_e, N_EDGES / 16);

    edge_max_kernel<<<gEH, THREADS, 0, stream>>>(src, dst, el, er, menc, indeg);
    edge_den_kernel<<<gEH, THREADS, 0, stream>>>(src, dst, el, er, menc, denom);
    aggregate_kernel<<<gEC, THREADS, 0, stream>>>(src, dst, el, er, menc, denom, hf, out);
    finalize_kernel<<<gNC, THREADS, 0, stream>>>(out, bias, hf, esum, indeg);
}
